// GraphSAGE_49615462203490
// MI455X (gfx1250) — matrix operand order repaired and checked by reading, not compiled, not run on hardware
//
#include <hip/hip_runtime.h>
#include <hip/hip_bf16.h>

typedef __attribute__((ext_vector_type(16))) __bf16 v16bf;
typedef __attribute__((ext_vector_type(8)))  float  v8f;

#define NN   100000
#define EE   1600000
#define DIN  128
#define HH   512
#define K1   256          // concatenated [x | mean] feature dim
#define K2   512
#define LDA  40           // padded LDS tile stride (bf16 elems): 80B -> conflict-free b128

#if defined(__has_builtin)
# if __has_builtin(__builtin_amdgcn_global_load_async_to_lds_b128)
#  define ASYNC_COPY 1
# endif
#endif

#if defined(ASYNC_COPY)
// Param 1 (per hipcc diagnostic): pointer to GCC-vector int4 in device/global AS.
// Param 2 (by symmetry with llvm.amdgcn.global.load.async.to.lds): same pointee in LDS AS3.
typedef int int4v __attribute__((vector_size(16)));
typedef __attribute__((address_space(1))) int4v* g_int4p;
typedef __attribute__((address_space(3))) int4v* l_int4p;

__device__ __forceinline__ void async_copy_b128(const void* g, void* l) {
    __builtin_amdgcn_global_load_async_to_lds_b128(
        (g_int4p)(unsigned long long)g,
        (l_int4p)(unsigned long long)l,   // flat LDS addr[31:0] == LDS offset (ISA aperture map)
        0, 0);
}
#endif

__device__ __forceinline__ void wait_async_le4() {
#if defined(ASYNC_COPY)
# if __has_builtin(__builtin_amdgcn_s_wait_asynccnt)
    __builtin_amdgcn_s_wait_asynccnt((short)4);
# else
    asm volatile("s_wait_asynccnt 0x4" ::: "memory");
# endif
#endif
}
__device__ __forceinline__ void wait_async_le0() {
#if defined(ASYNC_COPY)
# if __has_builtin(__builtin_amdgcn_s_wait_asynccnt)
    __builtin_amdgcn_s_wait_asynccnt((short)0);
# else
    asm volatile("s_wait_asynccnt 0x0" ::: "memory");
# endif
#endif
}

__device__ __forceinline__ __bf16 f2bf(float f) {
    union { float f; unsigned u; } in{f};
    unsigned r = in.u + 0x7FFFu + ((in.u >> 16) & 1u);   // round to nearest even
    unsigned short hv = (unsigned short)(r >> 16);
    return __builtin_bit_cast(__bf16, hv);
}

// ---------------- zero fill ----------------
__global__ __launch_bounds__(256) void zero_f32(float* p, long long n) {
    long long i = (long long)blockIdx.x * 256 + threadIdx.x;
    long long stride = (long long)gridDim.x * 256;
    for (; i < n; i += stride) p[i] = 0.0f;
}

// ---------------- edge scatter-add: one wave per edge ----------------
__global__ __launch_bounds__(256) void edge_agg(const float* __restrict__ x,
                                                const int* __restrict__ ei,
                                                float* __restrict__ agg,
                                                float* __restrict__ deg) {
    int e = blockIdx.x * 8 + (threadIdx.x >> 5);
    if (e >= EE) return;
    int lane = threadIdx.x & 31;
    int src = ei[e];
    int dst = ei[EE + e];
    const float4 v = *(const float4*)(x + (size_t)src * DIN + lane * 4);
    float* a = agg + (size_t)dst * DIN + lane * 4;
    atomicAdd(a + 0, v.x);
    atomicAdd(a + 1, v.y);
    atomicAdd(a + 2, v.z);
    atomicAdd(a + 3, v.w);
    if (lane == 0) atomicAdd(deg + dst, 1.0f);
}

// ---------------- pack [x | mean] to bf16 ----------------
__global__ __launch_bounds__(256) void pack_xmean(const float* __restrict__ x,
                                                  const float* __restrict__ agg,
                                                  const float* __restrict__ deg,
                                                  __bf16* __restrict__ xmean) {
    long long idx = (long long)blockIdx.x * 256 + threadIdx.x;   // over N*DIN
    if (idx >= (long long)NN * DIN) return;
    int row = (int)(idx >> 7);
    int col = (int)(idx & 127);
    float d = deg[row];
    float m = agg[idx] / fmaxf(d, 1.0f);
    xmean[(size_t)row * K1 + col]       = f2bf(x[idx]);
    xmean[(size_t)row * K1 + DIN + col] = f2bf(m);
}

// ---------------- pack weights, transposed (Wt[n][k]) ----------------
__global__ __launch_bounds__(256) void pack_weights(const float* __restrict__ W_l,
                                                    const float* __restrict__ W_r,
                                                    const float* __restrict__ W_proj,
                                                    __bf16* __restrict__ Wcat_t,   // [HH][K1]
                                                    __bf16* __restrict__ Wproj_t)  // [HH][K2]
{
    int idx = blockIdx.x * 256 + threadIdx.x;          // up to 512*512
    if (idx < HH * K1) {
        int n = idx >> 8;          // / K1
        int k = idx & (K1 - 1);
        float w = (k < DIN) ? W_r[(size_t)k * HH + n] : W_l[(size_t)(k - DIN) * HH + n];
        Wcat_t[idx] = f2bf(w);
    }
    if (idx < HH * K2) {
        int n = idx >> 9;          // / K2
        int k = idx & (K2 - 1);
        Wproj_t[idx] = f2bf(W_proj[(size_t)k * HH + n]);
    }
}

// ---------------- tiled bf16 WMMA GEMM ----------------
// C[M][Nc] = act( A[M][K] @ Bt[Nc][K]^T + bias )
// 256 threads = 8 waves; block tile 128x128, BK=32, double-buffered LDS.
// wave (wm in 0..3, wn in 0..1) owns a 32x64 sub-tile = 2x4 WMMA tiles.
union FragBF { v16bf v; float4 f4[2]; };

// Issue one stage's tile loads (4 x b128 per thread). OOB A rows are clamped:
// padding rows feed only epilogue-masked outputs.
__device__ __forceinline__ void stage_tiles(const __bf16* __restrict__ A,
                                            const __bf16* __restrict__ Bt,
                                            __bf16* Asm, __bf16* Bsm,
                                            int blockM, int blockN,
                                            int M, int K, int k0, int tid)
{
#pragma unroll
    for (int i = 0; i < 2; ++i) {
        int idx = tid + i * 256;        // 0..511 float4 slots per tile
        int tr  = idx >> 2;             // tile row 0..127
        int cs  = idx & 3;              // 8-element col segment
        int row = blockM + tr;
        row = (row < M) ? row : (M - 1);
        const __bf16* gA = A  + (size_t)row * K + k0 + cs * 8;
        __bf16*       lA = Asm + tr * LDA + cs * 8;
        int nrow = blockN + tr;         // Nc always a multiple of 128 here
        const __bf16* gB = Bt + (size_t)nrow * K + k0 + cs * 8;
        __bf16*       lB = Bsm + tr * LDA + cs * 8;
#if defined(ASYNC_COPY)
        async_copy_b128(gA, lA);
        async_copy_b128(gB, lB);
#else
        *(float4*)lA = *(const float4*)gA;
        *(float4*)lB = *(const float4*)gB;
#endif
    }
}

template <bool RELU, bool STORE_BF16>
__global__ __launch_bounds__(256) void gemm_wmma(const __bf16* __restrict__ A,
                                                 const __bf16* __restrict__ Bt,
                                                 const float*  __restrict__ bias,
                                                 float*        __restrict__ Cf,
                                                 __bf16*       __restrict__ Cb,
                                                 int M, int Nc, int K)
{
    __shared__ __align__(16) __bf16 Alds[2][128 * LDA];
    __shared__ __align__(16) __bf16 Blds[2][128 * LDA];

    const int tid  = threadIdx.x;
    const int lane = tid & 31;
    const int wave = tid >> 5;
    const int wm   = wave & 3;     // 0..3 : row group of 32
    const int wn   = wave >> 2;    // 0..1 : col group of 64
    const int r    = lane & 15;
    const int h    = lane >> 4;
    const int blockM = blockIdx.x * 128;
    const int blockN = blockIdx.y * 128;

    v8f acc[2][4];
#pragma unroll
    for (int m = 0; m < 2; ++m)
#pragma unroll
        for (int n = 0; n < 4; ++n)
            acc[m][n] = (v8f){0, 0, 0, 0, 0, 0, 0, 0};

    const int T = K >> 5;          // K / 32 stages
    stage_tiles(A, Bt, Alds[0], Blds[0], blockM, blockN, M, K, 0, tid);

    for (int kt = 0; kt < T; ++kt) {
        const int cur = kt & 1;
        if (kt + 1 < T) {
            stage_tiles(A, Bt, Alds[cur ^ 1], Blds[cur ^ 1],
                        blockM, blockN, M, K, (kt + 1) << 5, tid);
            wait_async_le4();      // the 4 in flight belong to next stage
        } else {
            wait_async_le0();
        }
        __syncthreads();           // stage `cur` visible to all waves

        const __bf16* Asm = Alds[cur];
        const __bf16* Bsm = Blds[cur];

        // A fragments (ISA 16-bit A layout: lane(r,h): K = h*8+j / 16+h*8+j)
        FragBF afr[2], bfr[4];
#pragma unroll
        for (int m = 0; m < 2; ++m) {
            int row = wm * 32 + m * 16 + r;
            afr[m].f4[0] = *(const float4*)(Asm + row * LDA + h * 8);
            afr[m].f4[1] = *(const float4*)(Asm + row * LDA + 16 + h * 8);
        }
        // B layout: lane(r,h): column n0+r, K = h*8+j / 16+h*8+j (same as A)
#pragma unroll
        for (int n = 0; n < 4; ++n) {
            int col = wn * 64 + n * 16 + r;
            bfr[n].f4[0] = *(const float4*)(Bsm + col * LDA + h * 8);
            bfr[n].f4[1] = *(const float4*)(Bsm + col * LDA + 16 + h * 8);
        }

#pragma unroll
        for (int m = 0; m < 2; ++m)
#pragma unroll
            for (int n = 0; n < 4; ++n)
                acc[m][n] = __builtin_amdgcn_wmma_f32_16x16x32_bf16(
                    false, afr[m].v, false, bfr[n].v,
                    (short)0, acc[m][n], false, false);

        __syncthreads();           // all waves done reading before next overwrite
    }

    // epilogue: D layout — element g: row = g + 8*h, col = r
#pragma unroll
    for (int m = 0; m < 2; ++m) {
#pragma unroll
        for (int n = 0; n < 4; ++n) {
            int colBase = blockN + wn * 64 + n * 16 + r;
            float bcol = bias[colBase];
#pragma unroll
            for (int g = 0; g < 8; ++g) {
                int row = blockM + wm * 32 + m * 16 + g + 8 * h;
                if (row < M) {
                    float v = acc[m][n][g] + bcol;
                    if (RELU) v = fmaxf(v, 0.0f);
                    if (STORE_BF16)
                        Cb[(size_t)row * Nc + colBase] = f2bf(v);
                    else
                        Cf[(size_t)row * Nc + colBase] = v;
                }
            }
        }
    }
}

extern "C" void kernel_launch(void* const* d_in, const int* in_sizes, int n_in,
                              void* d_out, int out_size, void* d_ws, size_t ws_size,
                              hipStream_t stream) {
    const float* x      = (const float*)d_in[0];
    const int*   ei     = (const int*)d_in[1];
    const float* W_l    = (const float*)d_in[2];
    const float* b_l    = (const float*)d_in[3];
    const float* W_r    = (const float*)d_in[4];
    const float* W_proj = (const float*)d_in[5];
    const float* b_proj = (const float*)d_in[6];
    float* out = (float*)d_out;

    // workspace carve-out (256B aligned)
    char* ws = (char*)d_ws;
    size_t off = 0;
    auto carve = [&](size_t bytes) -> void* {
        void* p = ws + off;
        off = (off + bytes + 255) & ~(size_t)255;
        return p;
    };
    float*   agg     = (float*)  carve((size_t)NN * DIN * 4);   // 51.2 MB
    float*   deg     = (float*)  carve((size_t)NN * 4);         // 0.4 MB
    __bf16*  xmean   = (__bf16*) carve((size_t)NN * K1 * 2);    // 51.2 MB
    __bf16*  hB      = (__bf16*) carve((size_t)NN * HH * 2);    // 102.4 MB
    __bf16*  Wcat_t  = (__bf16*) carve((size_t)HH * K1 * 2);    // 0.26 MB
    __bf16*  Wproj_t = (__bf16*) carve((size_t)HH * K2 * 2);    // 0.52 MB
    (void)ws_size;

    // 1) zero agg+deg (contiguous: NN*DIN*4 is a multiple of 256B)
    long long nz = (long long)NN * DIN + NN;
    zero_f32<<<4096, 256, 0, stream>>>(agg, nz);

    // 2) edge scatter-add (one wave per edge, 8 edges per block)
    edge_agg<<<(EE + 7) / 8, 256, 0, stream>>>(x, ei, agg, deg);

    // 3) pack [x | mean] as bf16
    pack_xmean<<<((long long)NN * DIN + 255) / 256, 256, 0, stream>>>(x, agg, deg, xmean);

    // 4) pack weights (transposed, bf16)
    pack_weights<<<(HH * K2) / 256, 256, 0, stream>>>(W_l, W_r, W_proj, Wcat_t, Wproj_t);

    // 5) h = relu(xmean @ Wcat + b_l)  -> bf16
    dim3 g1((NN + 127) / 128, HH / 128);
    gemm_wmma<true, true><<<g1, 256, 0, stream>>>(xmean, Wcat_t, b_l,
                                                  nullptr, hB, NN, HH, K1);

    // 6) out = h @ W_proj + b_proj    -> f32
    dim3 g2((NN + 127) / 128, HH / 128);
    gemm_wmma<false, false><<<g2, 256, 0, stream>>>(hB, Wproj_t, b_proj,
                                                    out, nullptr, NN, HH, K2);
}
